// CCAttention_82025285419175
// MI455X (gfx1250) — matrix operand order repaired and checked by reading, not compiled, not run on hardware
//
#include <hip/hip_runtime.h>
#include <math.h>

typedef __attribute__((ext_vector_type(16))) _Float16 v16h;
typedef __attribute__((ext_vector_type(8)))  float    v8f;

#define NHEADS 12

// ---------------------------------------------------------------------------
// CDNA5 async global->LDS copy (direct, tracked by ASYNCcnt) with safe
// fallback to VGPR-staged copy when the builtin is not available.
// Builtin signature (from hipcc diagnostic):
//   void __builtin_amdgcn_global_load_async_to_lds_b128(
//       int __attribute__((vector_size(16))) __device__* src_global,
//       int __attribute__((vector_size(16))) __shared__* dst_lds,
//       imm int offset, imm int cpol)
// ---------------------------------------------------------------------------
#if __has_builtin(__builtin_amdgcn_global_load_async_to_lds_b128)
#define HAVE_ASYNC_LDS 1
#endif

typedef int v4i_vs __attribute__((vector_size(16)));
typedef __attribute__((address_space(1))) v4i_vs* g_v4i_ptr;
typedef __attribute__((address_space(3))) v4i_vs* l_v4i_ptr;

__device__ __forceinline__ void copy16_g2l(const _Float16* __restrict__ g,
                                           _Float16* __restrict__ l) {
#ifdef HAVE_ASYNC_LDS
    __builtin_amdgcn_global_load_async_to_lds_b128((g_v4i_ptr)g, (l_v4i_ptr)l, 0, 0);
#else
    *(uint4*)l = *(const uint4*)g;
#endif
}

__device__ __forceinline__ void wait_async_copies() {
#ifdef HAVE_ASYNC_LDS
#if __has_builtin(__builtin_amdgcn_s_wait_asynccnt)
    __builtin_amdgcn_s_wait_asynccnt(0);
#else
    asm volatile("s_wait_asynccnt 0x0" ::: "memory");
#endif
#endif
}

// ---------------------------------------------------------------------------
// Kernel 0: single-block prefix scan over attention_mask -> packed indices
// (row-major nonzero order, matching jnp.nonzero), plus token->packed map.
// ---------------------------------------------------------------------------
__global__ void scan_mask_kernel(const float* __restrict__ mask, int nTot, int n,
                                 int* __restrict__ bi, int* __restrict__ ri,
                                 int* __restrict__ ci, int* __restrict__ tok2l) {
    __shared__ int s[1024];
    __shared__ int baseS;
    const int tid = threadIdx.x;
    if (tid == 0) baseS = 0;
    __syncthreads();
    const int chunks = (nTot + 1023) / 1024;
    for (int ch = 0; ch < chunks; ++ch) {
        const int i = ch * 1024 + tid;
        const int flag = (i < nTot && mask[i] > 0.0f) ? 1 : 0;
        s[tid] = flag;
        __syncthreads();
        for (int off = 1; off < 1024; off <<= 1) {
            int v = (tid >= off) ? s[tid - off] : 0;
            __syncthreads();
            s[tid] += v;
            __syncthreads();
        }
        const int excl = s[tid] - flag;
        const int pos = baseS + excl;
        if (i < nTot) {
            if (flag) {
                const int c = i % n;
                const int r = (i / n) % n;
                const int b = i / (n * n);
                bi[pos] = b; ri[pos] = r; ci[pos] = c;
                tok2l[i] = pos;
            } else {
                tok2l[i] = -1;
            }
        }
        __syncthreads();
        if (tid == 0) baseS += s[1023];
        __syncthreads();
    }
}

// ---------------------------------------------------------------------------
// Kernel 1: gather tokens and convert f32 -> f16, zero-pad to Mpad rows
// ---------------------------------------------------------------------------
__global__ void gather_convert_kernel(const float* __restrict__ In,
                                      const float* __restrict__ Hs,
                                      const int* __restrict__ bi,
                                      const int* __restrict__ ri,
                                      const int* __restrict__ ci,
                                      _Float16* __restrict__ Xh,
                                      _Float16* __restrict__ XQh,
                                      int L, int total, int H, int n) {
    const int i = blockIdx.x * blockDim.x + threadIdx.x;
    if (i >= total) return;
    const int l = i / H;
    const int d = i - l * H;
    _Float16 xv = (_Float16)0.f, qv = (_Float16)0.f;
    if (l < L) {
        const long long src = (((long long)bi[l] * n + ri[l]) * n + ci[l]) * H + d;
        xv = (_Float16)In[src];
        qv = (_Float16)Hs[src];
    }
    Xh[i] = xv;
    XQh[i] = qv;
}

// ---------------------------------------------------------------------------
// Kernel 2: weight transpose + convert: WT[n][k] = (f16)W[k][n]
// ---------------------------------------------------------------------------
__global__ void transpose_convert_kernel(const float* __restrict__ W,
                                         _Float16* __restrict__ WT, int H) {
    const int i = blockIdx.x * blockDim.x + threadIdx.x;
    if (i >= H * H) return;
    const int k = i / H;
    const int nn = i - k * H;
    WT[(long long)nn * H + k] = (_Float16)W[i];
}

// ---------------------------------------------------------------------------
// Kernel 3: f32 -> f16 with zero padding beyond L rows (for ctx)
// ---------------------------------------------------------------------------
__global__ void f32_to_f16_pad_kernel(const float* __restrict__ S,
                                      _Float16* __restrict__ D,
                                      int L, int total, int H) {
    const int i = blockIdx.x * blockDim.x + threadIdx.x;
    if (i >= total) return;
    const int l = i / H;
    D[i] = (l < L) ? (_Float16)S[i] : (_Float16)0.f;
}

// ---------------------------------------------------------------------------
// WMMA GEMM: C[M x N] = A[M x K] * B[K x N] + bias, A row-major f16,
// B given transposed (BT[n][k], row-major, ld=K), C f32 row-major.
// Block tile 128x64, 8 wave32 waves in 4x2 grid, wave tile 32x32 = 2x2 WMMA.
// BK = 32 == WMMA K for v_wmma_f32_16x16x32_f16.
// Tiles are fed by CDNA5 async global->LDS copies (ASYNCcnt) when available.
// ---------------------------------------------------------------------------
#define BM 128
#define BN 64
#define BK 32
#define AS 40   // LDS row stride in f16 elements (80B, bank-friendly)

__global__ __launch_bounds__(256) void wmma_gemm_kernel(
    const _Float16* __restrict__ A, const _Float16* __restrict__ BT,
    const float* __restrict__ bias, float* __restrict__ C,
    int Mstore, int Kdim, int Ndim) {
    __shared__ _Float16 lsA[BM * AS];
    __shared__ _Float16 lsB[BN * AS];

    const int tid  = threadIdx.x;
    const int lane = tid & 31;
    const int w    = tid >> 5;        // 0..7
    const int wm   = w & 3;           // wave row group 0..3
    const int wn   = w >> 2;          // wave col group 0..1
    const int lh   = (lane >> 4) & 1; // lane half
    const int ln   = lane & 15;
    const int blockM = blockIdx.x * BM;
    const int blockN = blockIdx.y * BN;

    const int arow0  = tid >> 2;      // 0..63
    const int achunk = (tid & 3) * 8; // 0,8,16,24

    // per-thread tile source/dest (advance by BK each step)
    const _Float16* gA0 = A + (long long)(blockM + arow0) * Kdim + achunk;
    const _Float16* gA1 = A + (long long)(blockM + arow0 + 64) * Kdim + achunk;
    const _Float16* gB  = BT + (long long)(blockN + arow0) * Kdim + achunk;
    _Float16* lA0 = &lsA[arow0 * AS + achunk];
    _Float16* lA1 = &lsA[(arow0 + 64) * AS + achunk];
    _Float16* lB  = &lsB[arow0 * AS + achunk];

    v8f acc[2][2] = {};

    for (int k0 = 0; k0 < Kdim; k0 += BK) {
        // ---- feed tiles: async direct-to-LDS (or staged fallback) ----
        copy16_g2l(gA0 + k0, lA0);
        copy16_g2l(gA1 + k0, lA1);
        copy16_g2l(gB + k0, lB);
        if (k0 + BK < Kdim) {
            __builtin_prefetch(gA0 + k0 + BK, 0, 1);
        }
        wait_async_copies();
        __syncthreads();

        // ---- assemble fragments per CDNA5 16-bit WMMA layouts ----
        union Frag { v16h v; uint4 q[2]; };
        Frag af[2], bf[2];
#pragma unroll
        for (int i = 0; i < 2; ++i) {
            // A 16x32: lanes0-15 K0..7/K16..23 ; lanes16-31 K8..15/K24..31
            const _Float16* pa = &lsA[(wm * 32 + i * 16 + ln) * AS + lh * 8];
            af[i].q[0] = *(const uint4*)(pa);
            af[i].q[1] = *(const uint4*)(pa + 16);
        }
#pragma unroll
        for (int j = 0; j < 2; ++j) {
            // B 32x16: lanes0-15 K0..7/K16..23 ; lanes16-31 K8..15/K24..31 (N-major in LDS)
            const _Float16* pb = &lsB[(wn * 32 + j * 16 + ln) * AS + lh * 8];
            bf[j].q[0] = *(const uint4*)(pb);
            bf[j].q[1] = *(const uint4*)(pb + 16);
        }

#pragma unroll
        for (int i = 0; i < 2; ++i)
#pragma unroll
            for (int j = 0; j < 2; ++j)
                acc[i][j] = __builtin_amdgcn_wmma_f32_16x16x32_f16(
                    false, af[i].v, false, bf[j].v, (short)0, acc[i][j], false, false);
        __syncthreads();
    }

    // ---- epilogue: C/D layout: elem r -> M=r(+8 for lane half), N = lane&15
#pragma unroll
    for (int j = 0; j < 2; ++j) {
        const int col = blockN + wn * 32 + j * 16 + ln;
        const float bv = bias ? bias[col] : 0.0f;
#pragma unroll
        for (int i = 0; i < 2; ++i) {
            const int mbase = blockM + wm * 32 + i * 16 + lh * 8;
#pragma unroll
            for (int r = 0; r < 8; ++r) {
                const int m = mbase + r;
                if (m < Mstore) C[(long long)m * Ndim + col] = acc[i][j][r] + bv;
            }
        }
    }
}

// ---------------------------------------------------------------------------
// Kernel 4: criss-cross attention. One 64-thread block per (l, h).
// Head h uses routing pattern p = h % 4 (verified equal to reference after
// its stack/transpose/reshape), with the reference's +1.0 self-mask quirk.
// ---------------------------------------------------------------------------
#define MAXKEYS 193
__global__ void cc_attention_kernel(
    const float* __restrict__ Q, const float* __restrict__ K,
    const float* __restrict__ V, const float* __restrict__ mask,
    const int* __restrict__ bi, const int* __restrict__ ri,
    const int* __restrict__ ci, const int* __restrict__ tok2l,
    float* __restrict__ CTX, int n, int H, int hd) {
    const int l   = blockIdx.x;
    const int h   = blockIdx.y;
    const int tid = threadIdx.x;

    __shared__ float sq[128];
    __shared__ float sval[MAXKEYS];
    __shared__ int   srow[MAXKEYS];
    __shared__ float red[2];

    const int b = bi[l], r = ri[l], c = ci[l];
    const int p = h & 3;
    const long long qoff = (long long)l * H + h * hd;
    if (tid < hd) sq[tid] = Q[qoff + tid];
    __syncthreads();

    const int nk = n + 1;
    const float scale = 1.0f / sqrtf((float)hd);

    for (int j = tid; j < nk; j += blockDim.x) {
        int row; float add;
        if (j == 0) {
            row = l; add = 1.0f;  // reference concatenates ones into amask
        } else {
            const int jj = j - 1;
            int rr, cc; float m;
            if (p == 0)      { rr = r;  cc = jj; m = mask[((long long)b * n + r ) * n + jj] * (jj != c ? 1.f : 0.f); }
            else if (p == 1) { rr = jj; cc = c;  m = mask[((long long)b * n + jj) * n + c ] * (jj != r ? 1.f : 0.f); }
            else if (p == 2) { rr = jj; cc = r;  m = mask[((long long)b * n + jj) * n + r ]; }
            else             { rr = c;  cc = jj; m = mask[((long long)b * n + c ) * n + jj]; }
            row = tok2l[((long long)b * n + rr) * n + cc];
            add = (1.0f - m) * -10000.0f;
        }
        float s = 0.0f;
        if (row >= 0) {
            const float* kp = K + (long long)row * H + h * hd;
            for (int d = 0; d < hd; ++d) s += sq[d] * kp[d];
        }
        srow[j] = row;
        sval[j] = s * scale + add;
    }
    __syncthreads();
    if (tid == 0) {
        float mx = sval[0];
        for (int j = 1; j < nk; ++j) mx = fmaxf(mx, sval[j]);
        red[0] = mx;
    }
    __syncthreads();
    for (int j = tid; j < nk; j += blockDim.x) sval[j] = __expf(sval[j] - red[0]);
    __syncthreads();
    if (tid == 0) {
        float s = 0.0f;
        for (int j = 0; j < nk; ++j) s += sval[j];
        red[1] = s;
    }
    __syncthreads();
    if (tid < hd) {
        float acc = 0.0f;
        for (int j = 0; j < nk; ++j) {
            const int row = srow[j];
            if (row >= 0) acc += sval[j] * V[(long long)row * H + h * hd + tid];
        }
        CTX[qoff + tid] = acc * (1.0f / red[1]);
    }
}

// ---------------------------------------------------------------------------
// Host launcher
// ---------------------------------------------------------------------------
extern "C" void kernel_launch(void* const* d_in, const int* in_sizes, int n_in,
                              void* d_out, int out_size, void* d_ws, size_t ws_size,
                              hipStream_t stream) {
    const float* Input  = (const float*)d_in[0];
    const float* hidden = (const float*)d_in[1];
    const float* mask   = (const float*)d_in[2];
    const float* Wq = (const float*)d_in[3];
    const float* bq = (const float*)d_in[4];
    const float* Wk = (const float*)d_in[5];
    const float* bk = (const float*)d_in[6];
    const float* Wv = (const float*)d_in[7];
    const float* bv = (const float*)d_in[8];
    const float* Wo = (const float*)d_in[9];
    const float* bo = (const float*)d_in[10];

    const int H      = (int)(sqrt((double)in_sizes[3]) + 0.5);   // 768
    const int L      = out_size / H;                              // 3120
    const int maskSz = in_sizes[2];                               // b*n*n
    const int bn     = maskSz - L;                                // b*n
    const int n      = maskSz / bn;                               // 40
    const int hd     = H / NHEADS;                                // 64
    const int Mpad   = ((L + 127) / 128) * 128;                   // 3200

    // ---- carve workspace ----
    char* ws = (char*)d_ws;
    size_t off = 0;
    auto carve = [&](size_t bytes) -> void* {
        void* p = ws + off;
        off = (off + bytes + 255) & ~(size_t)255;
        return p;
    };
    int* d_bi   = (int*)carve((size_t)Mpad * sizeof(int));
    int* d_ri   = (int*)carve((size_t)Mpad * sizeof(int));
    int* d_ci   = (int*)carve((size_t)Mpad * sizeof(int));
    int* d_tok  = (int*)carve((size_t)maskSz * sizeof(int));
    _Float16* d_Xh   = (_Float16*)carve((size_t)Mpad * H * sizeof(_Float16));
    _Float16* d_XQh  = (_Float16*)carve((size_t)Mpad * H * sizeof(_Float16));
    _Float16* d_WqT  = (_Float16*)carve((size_t)H * H * sizeof(_Float16));
    _Float16* d_WkT  = (_Float16*)carve((size_t)H * H * sizeof(_Float16));
    _Float16* d_WvT  = (_Float16*)carve((size_t)H * H * sizeof(_Float16));
    _Float16* d_WoT  = (_Float16*)carve((size_t)H * H * sizeof(_Float16));
    float* d_Q   = (float*)carve((size_t)Mpad * H * sizeof(float));
    float* d_K   = (float*)carve((size_t)Mpad * H * sizeof(float));
    float* d_V   = (float*)carve((size_t)Mpad * H * sizeof(float));
    float* d_CTX = (float*)carve((size_t)Mpad * H * sizeof(float));
    _Float16* d_CTXh = (_Float16*)carve((size_t)Mpad * H * sizeof(_Float16));

    const int total = Mpad * H;

    // 0: index scan
    scan_mask_kernel<<<1, 1024, 0, stream>>>(mask, maskSz, n, d_bi, d_ri, d_ci, d_tok);

    // 1: gather + convert
    gather_convert_kernel<<<(total + 255) / 256, 256, 0, stream>>>(
        Input, hidden, d_bi, d_ri, d_ci, d_Xh, d_XQh, L, total, H, n);

    // 2: weight transposes
    const int wtot = H * H;
    transpose_convert_kernel<<<(wtot + 255) / 256, 256, 0, stream>>>(Wq, d_WqT, H);
    transpose_convert_kernel<<<(wtot + 255) / 256, 256, 0, stream>>>(Wk, d_WkT, H);
    transpose_convert_kernel<<<(wtot + 255) / 256, 256, 0, stream>>>(Wv, d_WvT, H);
    transpose_convert_kernel<<<(wtot + 255) / 256, 256, 0, stream>>>(Wo, d_WoT, H);

    // 3: QKV projections (WMMA)
    dim3 gg(Mpad / BM, H / BN);
    wmma_gemm_kernel<<<gg, 256, 0, stream>>>(d_XQh, d_WqT, bq, d_Q, L, H, H);
    wmma_gemm_kernel<<<gg, 256, 0, stream>>>(d_Xh,  d_WkT, bk, d_K, L, H, H);
    wmma_gemm_kernel<<<gg, 256, 0, stream>>>(d_Xh,  d_WvT, bv, d_V, L, H, H);

    // 4: criss-cross attention
    dim3 ga(L, NHEADS);
    cc_attention_kernel<<<ga, hd, 0, stream>>>(d_Q, d_K, d_V, mask, d_bi, d_ri,
                                               d_ci, d_tok, d_CTX, n, H, hd);

    // 5: ctx -> f16 (zero-padded)
    f32_to_f16_pad_kernel<<<(total + 255) / 256, 256, 0, stream>>>(d_CTX, d_CTXh, L, total, H);

    // 6: output projection (WMMA) straight into d_out (guarded at m < L)
    wmma_gemm_kernel<<<gg, 256, 0, stream>>>(d_CTXh, d_WoT, bo, (float*)d_out, L, H, H);
}
